// Model_53901839565540
// MI455X (gfx1250) — matrix operand order repaired and checked by reading, not compiled, not run on hardware
//
#include <hip/hip_runtime.h>

// ---------------------------------------------------------------------------
// Types for CDNA5 WMMA (wave32)
// ---------------------------------------------------------------------------
typedef __attribute__((ext_vector_type(16))) __bf16 v16bf;
typedef __attribute__((ext_vector_type(8)))  float  v8f;

struct Pack32 { uint4 lo, hi; };              // 32 bytes -> v16bf
struct P16   { unsigned short u[16]; };       // 32 bytes -> v16bf

__device__ __forceinline__ unsigned short f2bf(float f) {
    unsigned int u = __builtin_bit_cast(unsigned int, f);
    unsigned int r = u + 0x7FFFu + ((u >> 16) & 1u);   // RNE
    return (unsigned short)(r >> 16);
}
__device__ __forceinline__ float bf2f(unsigned short s) {
    unsigned int u = ((unsigned int)s) << 16;
    return __builtin_bit_cast(float, u);
}

// ---------------------------------------------------------------------------
// Small helper kernels
// ---------------------------------------------------------------------------
__global__ void zero_f32_kernel(float* __restrict__ p, int n) {
    int i = blockIdx.x * blockDim.x + threadIdx.x;
    if (i < n) p[i] = 0.f;
}

__global__ void f32_to_bf16_kernel(const float* __restrict__ s,
                                   unsigned short* __restrict__ d, int n) {
    int i = blockIdx.x * blockDim.x + threadIdx.x;
    if (i < n) d[i] = f2bf(s[i]);
}

__global__ void count_kernel(const int* __restrict__ dst,
                             float* __restrict__ cnt, int E) {
    int i = blockIdx.x * blockDim.x + threadIdx.x;
    if (i < E)
        __hip_atomic_fetch_add(&cnt[dst[i]], 1.0f,
                               __ATOMIC_RELAXED, __HIP_MEMORY_SCOPE_AGENT);
}

__global__ void invcnt_kernel(const float* __restrict__ cnt,
                              float* __restrict__ inv, int n) {
    int i = blockIdx.x * blockDim.x + threadIdx.x;
    if (i < n) inv[i] = 1.0f / fmaxf(cnt[i], 1.0f);
}

// Segment-sum of bf16 node features into f32 agg buffer (4 features / thread)
__global__ void gather_kernel(const int* __restrict__ src,
                              const int* __restrict__ dst,
                              const unsigned short* __restrict__ x_bf,
                              float* __restrict__ agg, int E) {
    int tid = blockIdx.x * blockDim.x + threadIdx.x;
    int total = E * 24;                 // 96 feats / 4 per thread
    if (tid >= total) return;
    int e = tid / 24;
    int c = (tid % 24) * 4;
    int s = src[e], d = dst[e];
    const unsigned short* xp = x_bf + (size_t)s * 96 + c;
    float* ap = agg + (size_t)d * 96 + c;
    #pragma unroll
    for (int i = 0; i < 4; ++i)
        __hip_atomic_fetch_add(ap + i, bf2f(xp[i]),
                               __ATOMIC_RELAXED, __HIP_MEMORY_SCOPE_AGENT);
}

// Embedding gather + [80 -> 96] linear, output bf16. One node per 96-thr block.
__global__ __launch_bounds__(96) void embed_kernel(
    const int* __restrict__ x_layout, const int* __restrict__ x_role,
    const float* __restrict__ layout_emb, const float* __restrict__ role_emb,
    const float* __restrict__ W, const float* __restrict__ b,
    unsigned short* __restrict__ out_bf, int N) {
    __shared__ float feat[80];
    int node = blockIdx.x;
    if (node >= N) return;
    int t = threadIdx.x;
    if (t < 64) {
        int slot = t >> 4, dim = t & 15;
        feat[t] = layout_emb[x_layout[node * 4 + slot] * 16 + dim];
    } else if (t < 80) {
        feat[t] = role_emb[x_role[node] * 16 + (t - 64)];
    }
    __syncthreads();
    float acc = b[t];
    #pragma unroll 8
    for (int k = 0; k < 80; ++k) acc += W[t * 80 + k] * feat[k];
    out_bf[(size_t)node * 96 + t] = f2bf(acc);
}

// ---------------------------------------------------------------------------
// Fused SAGE layer: out = relu(norm(agg) @ lW^T + lb + x @ rW^T)  (all 96-dim)
// One wave computes a 16x96 output stripe: 6 f32 accumulators, K = 3x32 bf16.
// ---------------------------------------------------------------------------
__global__ __launch_bounds__(256) void sage_wmma_kernel(
    const unsigned short* __restrict__ x_bf,   // [N,96] bf16
    const float*          __restrict__ agg,    // [N,96] f32
    const float*          __restrict__ invcnt, // [N]
    const unsigned short* __restrict__ lW,     // [96,96] bf16 (row = out)
    const unsigned short* __restrict__ rW,     // [96,96] bf16
    const float*          __restrict__ lb,     // [96]
    unsigned short*       __restrict__ out_bf, // [N,96] bf16
    int N, int nRowTiles) {
    const int wave = threadIdx.x >> 5;
    const int tile = blockIdx.x * 8 + wave;
    if (tile >= nRowTiles) return;
    const int lane = threadIdx.x & 31;
    const int half = lane >> 4;
    const int mn   = lane & 15;
    int row = tile * 16 + mn;
    if (row >= N) row = N - 1;              // safe clamp (N % 16 == 0 normally)
    const float inv = invcnt[row];

    v8f acc[6];
    #pragma unroll
    for (int t = 0; t < 6; ++t) {
        float bv = lb[t * 16 + mn];
        #pragma unroll
        for (int r = 0; r < 8; ++r) acc[t][r] = bv;
    }

    #pragma unroll
    for (int ks = 0; ks < 3; ++ks) {
        const int k0 = ks * 32;
        // A operand from x (bf16): chunks K = k0+half*8 .. +7 and k0+16+half*8 .. +7
        Pack32 ax;
        ax.lo = *(const uint4*)(x_bf + (size_t)row * 96 + k0 + half * 8);
        ax.hi = *(const uint4*)(x_bf + (size_t)row * 96 + k0 + 16 + half * 8);
        v16bf a_x = __builtin_bit_cast(v16bf, ax);
        // A operand from normalized agg (f32 -> bf16 on the fly)
        P16 ag;
        const float* ap0 = agg + (size_t)row * 96 + k0 + half * 8;
        const float* ap1 = agg + (size_t)row * 96 + k0 + 16 + half * 8;
        #pragma unroll
        for (int i = 0; i < 8; ++i) ag.u[i]     = f2bf(ap0[i] * inv);
        #pragma unroll
        for (int i = 0; i < 8; ++i) ag.u[8 + i] = f2bf(ap1[i] * inv);
        v16bf a_g = __builtin_bit_cast(v16bf, ag);

        #pragma unroll
        for (int t = 0; t < 6; ++t) {
            const int col = t * 16 + mn;
            Pack32 bl;
            const unsigned short* pl = lW + (size_t)col * 96 + k0 + half * 8;
            bl.lo = *(const uint4*)(pl);
            bl.hi = *(const uint4*)(pl + 16);
            v16bf b_l = __builtin_bit_cast(v16bf, bl);
            acc[t] = __builtin_amdgcn_wmma_f32_16x16x32_bf16(
                false, a_g, false, b_l, (short)0, acc[t], false, false);
            Pack32 br;
            const unsigned short* pr = rW + (size_t)col * 96 + k0 + half * 8;
            br.lo = *(const uint4*)(pr);
            br.hi = *(const uint4*)(pr + 16);
            v16bf b_r = __builtin_bit_cast(v16bf, br);
            acc[t] = __builtin_amdgcn_wmma_f32_16x16x32_bf16(
                false, a_x, false, b_r, (short)0, acc[t], false, false);
        }
    }

    // ReLU + store bf16. D row = half*8 + r, col = t*16 + mn.
    #pragma unroll
    for (int t = 0; t < 6; ++t) {
        #pragma unroll
        for (int r = 0; r < 8; ++r) {
            int orow = tile * 16 + half * 8 + r;
            if (orow < N) {
                float v = acc[t][r];
                v = v > 0.f ? v : 0.f;
                out_bf[(size_t)orow * 96 + t * 16 + mn] = f2bf(v);
            }
        }
    }
}

// ---------------------------------------------------------------------------
// Generic dense layer: out = relu(in @ W^T + b), in [N,K] bf16, W [OUT,K] bf16.
// One wave per 16x16 output tile. K, OUT multiples of 32 / 16.
// ---------------------------------------------------------------------------
__global__ __launch_bounds__(256) void dense_wmma_kernel(
    const unsigned short* __restrict__ in_bf,
    const unsigned short* __restrict__ W,
    const float*          __restrict__ b,
    unsigned short*       __restrict__ out_bf,
    int K, int OUT, int N, int nRowTiles) {
    const int colTiles = OUT >> 4;
    const int wid = blockIdx.x * 8 + (threadIdx.x >> 5);
    if (wid >= nRowTiles * colTiles) return;
    const int rowTile = wid / colTiles;
    const int colTile = wid % colTiles;
    const int lane = threadIdx.x & 31;
    const int half = lane >> 4;
    const int mn   = lane & 15;
    int row = rowTile * 16 + mn;
    if (row >= N) row = N - 1;
    const int col = colTile * 16 + mn;

    v8f acc;
    {
        float bv = b[col];
        #pragma unroll
        for (int r = 0; r < 8; ++r) acc[r] = bv;
    }

    for (int k0 = 0; k0 < K; k0 += 32) {
        Pack32 ax;
        const unsigned short* xp = in_bf + (size_t)row * K + k0;
        ax.lo = *(const uint4*)(xp + half * 8);
        ax.hi = *(const uint4*)(xp + 16 + half * 8);
        v16bf a = __builtin_bit_cast(v16bf, ax);
        Pack32 bw;
        const unsigned short* wp = W + (size_t)col * K + k0 + half * 8;
        bw.lo = *(const uint4*)(wp);
        bw.hi = *(const uint4*)(wp + 16);
        v16bf bb = __builtin_bit_cast(v16bf, bw);
        acc = __builtin_amdgcn_wmma_f32_16x16x32_bf16(
            false, a, false, bb, (short)0, acc, false, false);
    }

    #pragma unroll
    for (int r = 0; r < 8; ++r) {
        int orow = rowTile * 16 + half * 8 + r;
        if (orow < N) {
            float v = acc[r];
            v = v > 0.f ? v : 0.f;
            out_bf[(size_t)orow * OUT + col] = f2bf(v);
        }
    }
}

// Final 256 -> 1 projection: one wave per node, shuffle reduction (wave32).
__global__ void head_kernel(const unsigned short* __restrict__ h,
                            const float* __restrict__ w,
                            const float* __restrict__ b,
                            float* __restrict__ out, int N) {
    int wid = (blockIdx.x * blockDim.x + threadIdx.x) >> 5;
    int lane = threadIdx.x & 31;
    if (wid >= N) return;
    float s = 0.f;
    #pragma unroll
    for (int k = lane; k < 256; k += 32)
        s += bf2f(h[(size_t)wid * 256 + k]) * w[k];
    #pragma unroll
    for (int off = 16; off > 0; off >>= 1)
        s += __shfl_down(s, off, 32);
    if (lane == 0) out[wid] = s + b[0];
}

// ---------------------------------------------------------------------------
// Host launch
// ---------------------------------------------------------------------------
extern "C" void kernel_launch(void* const* d_in, const int* in_sizes, int n_in,
                              void* d_out, int out_size, void* d_ws, size_t ws_size,
                              hipStream_t stream) {
    const int*   x_layout   = (const int*)d_in[0];
    const int*   x_role     = (const int*)d_in[1];
    const int*   edge       = (const int*)d_in[2];
    const float* layout_emb = (const float*)d_in[3];
    const float* role_emb   = (const float*)d_in[4];
    const float* lin_W      = (const float*)d_in[5];
    const float* lin_b      = (const float*)d_in[6];
    const float* c_lW[3] = {(const float*)d_in[7],  (const float*)d_in[10], (const float*)d_in[13]};
    const float* c_lb[3] = {(const float*)d_in[8],  (const float*)d_in[11], (const float*)d_in[14]};
    const float* c_rW[3] = {(const float*)d_in[9],  (const float*)d_in[12], (const float*)d_in[15]};
    const float* d0_W = (const float*)d_in[16];
    const float* d0_b = (const float*)d_in[17];
    const float* d1_W = (const float*)d_in[18];
    const float* d1_b = (const float*)d_in[19];
    const float* d2_W = (const float*)d_in[20];
    const float* d2_b = (const float*)d_in[21];

    const int N = in_sizes[1];        // x_role count = #nodes
    const int E = in_sizes[2] / 2;    // edge_index is [2, E]
    const int* src = edge;
    const int* dst = edge + E;

    // ---- workspace carve-out (256B aligned slices) ----
    size_t off = 0;
    auto take = [&](size_t bytes) -> char* {
        char* p = (char*)d_ws + off;
        off += (bytes + 255) & ~(size_t)255;
        return p;
    };
    unsigned short* xA   = (unsigned short*)take((size_t)N * 96 * 2);
    unsigned short* xB   = (unsigned short*)take((size_t)N * 96 * 2);
    unsigned short* h0   = (unsigned short*)take((size_t)N * 256 * 2);
    unsigned short* h1   = (unsigned short*)take((size_t)N * 256 * 2);
    float*          agg  = (float*)take((size_t)N * 96 * 4);
    float*          cnt  = (float*)take((size_t)N * 4);
    float*          inv  = (float*)take((size_t)N * 4);
    unsigned short* lWbf[3], *rWbf[3];
    for (int l = 0; l < 3; ++l) {
        lWbf[l] = (unsigned short*)take(96 * 96 * 2);
        rWbf[l] = (unsigned short*)take(96 * 96 * 2);
    }
    unsigned short* d0Wbf = (unsigned short*)take(256 * 96 * 2);
    unsigned short* d1Wbf = (unsigned short*)take((size_t)256 * 256 * 2);
    (void)ws_size; (void)n_in; (void)out_size;

    const int TPB = 256;
    auto blocks = [&](long long n) { return (int)((n + TPB - 1) / TPB); };

    // ---- weight down-conversion to bf16 ----
    for (int l = 0; l < 3; ++l) {
        f32_to_bf16_kernel<<<blocks(96 * 96), TPB, 0, stream>>>(c_lW[l], lWbf[l], 96 * 96);
        f32_to_bf16_kernel<<<blocks(96 * 96), TPB, 0, stream>>>(c_rW[l], rWbf[l], 96 * 96);
    }
    f32_to_bf16_kernel<<<blocks(256 * 96), TPB, 0, stream>>>(d0_W, d0Wbf, 256 * 96);
    f32_to_bf16_kernel<<<blocks(256 * 256), TPB, 0, stream>>>(d1_W, d1Wbf, 256 * 256);

    // ---- in-degree reciprocal ----
    zero_f32_kernel<<<blocks(N), TPB, 0, stream>>>(cnt, N);
    count_kernel<<<blocks(E), TPB, 0, stream>>>(dst, cnt, E);
    invcnt_kernel<<<blocks(N), TPB, 0, stream>>>(cnt, inv, N);

    // ---- embeddings + input linear ----
    embed_kernel<<<N, 96, 0, stream>>>(x_layout, x_role, layout_emb, role_emb,
                                       lin_W, lin_b, xA, N);

    // ---- three fused SAGE layers ----
    const int nRowTiles = (N + 15) / 16;
    const int sageGrid  = (nRowTiles + 7) / 8;
    unsigned short* xin = xA;
    unsigned short* xout = xB;
    for (int l = 0; l < 3; ++l) {
        zero_f32_kernel<<<blocks((long long)N * 96), TPB, 0, stream>>>(agg, N * 96);
        gather_kernel<<<blocks((long long)E * 24), TPB, 0, stream>>>(src, dst, xin, agg, E);
        sage_wmma_kernel<<<sageGrid, TPB, 0, stream>>>(xin, agg, inv,
                                                       lWbf[l], rWbf[l], c_lb[l],
                                                       xout, N, nRowTiles);
        unsigned short* t = xin; xin = xout; xout = t;
    }
    // after 3 swaps, xin holds final graph features [N,96] bf16

    // ---- dense head ----
    {
        int colTiles = 256 / 16;
        int totalWaves = nRowTiles * colTiles;
        int grid = (totalWaves + 7) / 8;
        dense_wmma_kernel<<<grid, TPB, 0, stream>>>(xin, d0Wbf, d0_b, h0, 96, 256, N, nRowTiles);
        dense_wmma_kernel<<<grid, TPB, 0, stream>>>(h0, d1Wbf, d1_b, h1, 256, 256, N, nRowTiles);
    }
    head_kernel<<<blocks((long long)N * 32), TPB, 0, stream>>>(h1, d2_W, d2_b, (float*)d_out, N);
}
